// GraphTransformer_19945828122967
// MI455X (gfx1250) — compile-verified, matrix operand order checked by reading, not run on hardware
//
#include <hip/hip_runtime.h>
#include <cmath>

// ---------------------------------------------------------------------------
// Types for CDNA5 WMMA (wave32, 16x16x32 bf16 -> f32)
// ---------------------------------------------------------------------------
typedef __attribute__((ext_vector_type(16))) __bf16 v16bf;
typedef __attribute__((ext_vector_type(8)))  float  v8f;
typedef __attribute__((ext_vector_type(4)))  int    vi4;

// bf16 encode (round-to-nearest-even, ignoring NaN corner)
__device__ __forceinline__ unsigned short f2bf(float f) {
    unsigned u = __float_as_uint(f);
    u += 0x7FFFu + ((u >> 16) & 1u);
    return (unsigned short)(u >> 16);
}
__device__ __forceinline__ float bf2f(unsigned short h) {
    return __uint_as_float((unsigned)h << 16);
}

// float atomic max via sign-aware integer atomics (init with -inf)
__device__ __forceinline__ void atomicMaxFloat(float* addr, float val) {
    if (val >= 0.0f)
        atomicMax((int*)addr, __float_as_int(val));
    else
        atomicMin((unsigned int*)addr, __float_as_uint(val));
}

// ---------------------------------------------------------------------------
// Utility kernels
// ---------------------------------------------------------------------------
__global__ void fill_f32(float* __restrict__ p, float v, size_t n) {
    size_t i = (size_t)blockIdx.x * blockDim.x + threadIdx.x;
    if (i < n) p[i] = v;
}

// fp32 [nvalid] -> bf16 hi/lo split [ntotal], zero padding past nvalid
__global__ void cvt_pad_bf16x2(const float* __restrict__ src,
                               unsigned short* __restrict__ hi,
                               unsigned short* __restrict__ lo,
                               size_t nvalid, size_t ntotal) {
    size_t i = (size_t)blockIdx.x * blockDim.x + threadIdx.x;
    if (i >= ntotal) return;
    const float f = (i < nvalid) ? src[i] : 0.0f;
    const unsigned short h = f2bf(f);
    hi[i] = h;
    lo[i] = f2bf(f - bf2f(h));
}

// W [K x M] fp32 (row-major) -> WT_hi/WT_lo [M x K] bf16 (row-major)
__global__ void transpose_cvt_w_split(const float* __restrict__ W,
                                      unsigned short* __restrict__ WThi,
                                      unsigned short* __restrict__ WTlo,
                                      int K, int M) {
    size_t i = (size_t)blockIdx.x * blockDim.x + threadIdx.x;
    if (i >= (size_t)K * M) return;
    int m = (int)(i / K);
    int k = (int)(i - (size_t)m * K);
    const float f = W[(size_t)k * M + m];
    const unsigned short h = f2bf(f);
    WThi[i] = h;
    WTlo[i] = f2bf(f - bf2f(h));
}

// ---------------------------------------------------------------------------
// bf16x3 WMMA GEMM (fp32-class accuracy):
//   Y[Npad x M] = A[Npad x K] * BT[M x K]^T + bias,
//   A = Ahi + Alo, B = Bhi + Blo; accumulate hi*hi + hi*lo + lo*hi in f32.
// Block: 128 threads = 4 waves. Block tile: 16 rows x 256 cols.
// Wave tile: 16 x 64 (4 accumulators). A hi/lo tiles staged via LDS.
// ---------------------------------------------------------------------------
__global__ __launch_bounds__(128)
void gemm_bf16x3_wmma(const unsigned short* __restrict__ Ahi,
                      const unsigned short* __restrict__ Alo,
                      const unsigned short* __restrict__ Bhi,
                      const unsigned short* __restrict__ Blo,
                      const float* __restrict__ bias,
                      float* __restrict__ Y,
                      int K, int M) {
    __shared__ __align__(16) unsigned long long sA64[256]; // hi tile (1KB) + lo tile (1KB)

    const int tid  = threadIdx.x;
    const int wave = tid >> 5;
    const int lane = tid & 31;
    const int l15  = lane & 15;
    const int half = lane >> 4;

    const int mrow = blockIdx.y * 16;              // node-row tile base
    const int ncol = blockIdx.x * 256 + wave * 64; // wave column base

    v8f acc[4] = {v8f{}, v8f{}, v8f{}, v8f{}};

    const int ar = tid >> 3;            // staging row 0..15
    const int ac = (tid & 7) * 4;       // staging element offset (4 bf16 = 8B)
    const char* sAb = (const char*)sA64;

    for (int kk = 0; kk < K; kk += 32) {
        // Cooperative A-tile stage: hi then lo, 128 threads x 8B each
        const size_t aoff = (size_t)(mrow + ar) * K + kk + ac;
        sA64[tid]       = *(const unsigned long long*)(Ahi + aoff);
        sA64[128 + tid] = *(const unsigned long long*)(Alo + aoff);
        if (kk + 32 < K) {
            __builtin_prefetch(Ahi + aoff + 32, 0, 1);
            __builtin_prefetch(Alo + aoff + 32, 0, 1);
        }
        __syncthreads();

        // A fragments (16x32 bf16): two b128 LDS reads per lane, hi and lo
        v16bf afh, afl;
        ((vi4*)&afh)[0] = *(const vi4*)(sAb + l15 * 64 + half * 16);
        ((vi4*)&afh)[1] = *(const vi4*)(sAb + l15 * 64 + 32 + half * 16);
        ((vi4*)&afl)[0] = *(const vi4*)(sAb + 1024 + l15 * 64 + half * 16);
        ((vi4*)&afl)[1] = *(const vi4*)(sAb + 1024 + l15 * 64 + 32 + half * 16);

        // 4 N-tiles; B hi/lo fragments streamed from global (rows contiguous in K)
        #pragma unroll
        for (int t = 0; t < 4; ++t) {
            const size_t boff = (size_t)(ncol + t * 16 + l15) * K + kk;
            const unsigned short* bph = Bhi + boff;
            const unsigned short* bpl = Blo + boff;
            v16bf bfh, bfl;
            ((vi4*)&bfh)[0] = *(const vi4*)(bph + half * 8);
            ((vi4*)&bfh)[1] = *(const vi4*)(bph + 16 + half * 8);
            ((vi4*)&bfl)[0] = *(const vi4*)(bpl + half * 8);
            ((vi4*)&bfl)[1] = *(const vi4*)(bpl + 16 + half * 8);
            acc[t] = __builtin_amdgcn_wmma_f32_16x16x32_bf16(false, afh, false, bfh, (short)0, acc[t], false, false);
            acc[t] = __builtin_amdgcn_wmma_f32_16x16x32_bf16(false, afh, false, bfl, (short)0, acc[t], false, false);
            acc[t] = __builtin_amdgcn_wmma_f32_16x16x32_bf16(false, afl, false, bfh, (short)0, acc[t], false, false);
        }
        __syncthreads();
    }

    // Epilogue: C/D layout — VGPR r: lanes0-15 M=r, lanes16-31 M=r+8; N = lane&15
    #pragma unroll
    for (int t = 0; t < 4; ++t) {
        const int n = ncol + t * 16 + l15;
        const float bv = bias[n];
        #pragma unroll
        for (int r = 0; r < 8; ++r) {
            const int row = mrow + r + half * 8;
            Y[(size_t)row * M + n] = acc[t][r] + bv;
        }
    }
}

// ---------------------------------------------------------------------------
// Edge attention kernels (H = 4 fixed)
// ---------------------------------------------------------------------------
// alpha[e,h] = dot(q[dst[e],h,:], k[src[e],h,:]) * inv_scale ; one wave per (e,h)
__global__ __launch_bounds__(128)
void edge_logits(const int* __restrict__ src, const int* __restrict__ dst,
                 const float* __restrict__ q, const float* __restrict__ k,
                 float* __restrict__ alpha, int EH, int C, float inv_scale) {
    const int idx  = blockIdx.x * 4 + (threadIdx.x >> 5);
    const int lane = threadIdx.x & 31;
    if (idx >= EH) return;
    const int e = idx >> 2, h = idx & 3;
    const int sN = src[e], dN = dst[e];
    const int HC = 4 * C;
    const float* qp = q + (size_t)dN * HC + h * C;
    const float* kp = k + (size_t)sN * HC + h * C;
    float acc = 0.0f;
    for (int c = lane; c < C; c += 32) acc += qp[c] * kp[c];
    #pragma unroll
    for (int o = 16; o > 0; o >>= 1) acc += __shfl_xor(acc, o, 32);
    if (lane == 0) alpha[idx] = acc * inv_scale;
}

__global__ void seg_max(const int* __restrict__ dst, const float* __restrict__ alpha,
                        float* __restrict__ m, int EH) {
    int i = blockIdx.x * blockDim.x + threadIdx.x;
    if (i >= EH) return;
    atomicMaxFloat(&m[dst[i >> 2] * 4 + (i & 3)], alpha[i]);
}

__global__ void exp_seg_sum(const int* __restrict__ dst, float* __restrict__ alpha,
                            const float* __restrict__ m, float* __restrict__ s, int EH) {
    int i = blockIdx.x * blockDim.x + threadIdx.x;
    if (i >= EH) return;
    const int mi = dst[i >> 2] * 4 + (i & 3);
    const float a = __expf(alpha[i] - m[mi]);
    alpha[i] = a;
    atomicAdd(&s[mi], a);
}

// out[dst,h,:] += v[src,h,:] * (a / max(s,1)) ; one wave per (e,h)
__global__ __launch_bounds__(128)
void scatter_v(const int* __restrict__ src, const int* __restrict__ dst,
               const float* __restrict__ v, const float* __restrict__ alpha,
               const float* __restrict__ s, float* __restrict__ out,
               int EH, int C) {
    const int idx  = blockIdx.x * 4 + (threadIdx.x >> 5);
    const int lane = threadIdx.x & 31;
    if (idx >= EH) return;
    const int e = idx >> 2, h = idx & 3;
    const int sN = src[e], dN = dst[e];
    const float ss = s[dN * 4 + h];
    const float w  = alpha[idx] / (ss > 0.0f ? ss : 1.0f);
    const int HC = 4 * C;
    const float* vp = v + (size_t)sN * HC + h * C;
    float* op = out + (size_t)dN * HC + h * C;
    for (int c = lane; c < C; c += 32) atomicAdd(&op[c], vp[c] * w);
}

// h = relu(agg + root), in place into agg
__global__ void add_root_relu(float* __restrict__ agg, const float* __restrict__ r, size_t n) {
    size_t i = (size_t)blockIdx.x * blockDim.x + threadIdx.x;
    if (i >= n) return;
    const float o = agg[i] + r[i];
    agg[i] = o > 0.0f ? o : 0.0f;
}

// ---------------------------------------------------------------------------
// Pooling + final linear
// ---------------------------------------------------------------------------
__global__ void pool_cnt(const int* __restrict__ batch, float* __restrict__ cnt, int N) {
    int i = blockIdx.x * blockDim.x + threadIdx.x;
    if (i < N) atomicAdd(&cnt[batch[i]], 1.0f);
}

__global__ void pool_sum(const int* __restrict__ batch, const float* __restrict__ h,
                         float* __restrict__ gsum, int N, int D) {
    size_t i = (size_t)blockIdx.x * blockDim.x + threadIdx.x;
    if (i >= (size_t)N * D) return;
    const int n = (int)(i / D);
    atomicAdd(&gsum[(size_t)batch[n] * D + (i - (size_t)n * D)], h[i]);
}

__global__ __launch_bounds__(256)
void pool_linear(const float* __restrict__ gsum, const float* __restrict__ cnt,
                 const float* __restrict__ Wlin, const float* __restrict__ blin,
                 float* __restrict__ out, int D) {
    __shared__ float red[256];
    const int g = blockIdx.x;
    const float inv = 1.0f / fmaxf(cnt[g], 1.0f);
    float acc = 0.0f;
    for (int i = threadIdx.x; i < D; i += 256) acc += gsum[(size_t)g * D + i] * Wlin[i];
    red[threadIdx.x] = acc * inv;
    __syncthreads();
    for (int st = 128; st > 0; st >>= 1) {
        if (threadIdx.x < st) red[threadIdx.x] += red[threadIdx.x + st];
        __syncthreads();
    }
    if (threadIdx.x == 0) out[g] = red[0] + blin[0];
}

// ---------------------------------------------------------------------------
// Host orchestration
// ---------------------------------------------------------------------------
extern "C" void kernel_launch(void* const* d_in, const int* in_sizes, int n_in,
                              void* d_out, int out_size, void* d_ws, size_t ws_size,
                              hipStream_t stream) {
    const int F0 = 256, D1 = 512, D2 = 1024;
    const int N  = in_sizes[0] / F0;       // 10000
    const int E  = in_sizes[1] / 2;        // 160000
    const int G  = out_size;               // 16
    const int EH = E * 4;
    const int NPAD = ((N + 15) / 16) * 16; // 10000

    const float* x   = (const float*)d_in[0];
    const int*   src = (const int*)d_in[1];
    const int*   dst = src + E;
    const int*   batch = (const int*)d_in[3];
    const float* Wq[2] = { (const float*)d_in[5],  (const float*)d_in[13] };
    const float* bq[2] = { (const float*)d_in[6],  (const float*)d_in[14] };
    const float* Wk[2] = { (const float*)d_in[7],  (const float*)d_in[15] };
    const float* bk[2] = { (const float*)d_in[8],  (const float*)d_in[16] };
    const float* Wv[2] = { (const float*)d_in[9],  (const float*)d_in[17] };
    const float* bv[2] = { (const float*)d_in[10], (const float*)d_in[18] };
    const float* Wr[2] = { (const float*)d_in[11], (const float*)d_in[19] };
    const float* br[2] = { (const float*)d_in[12], (const float*)d_in[20] };
    const float* Wlin = (const float*)d_in[21];
    const float* blin = (const float*)d_in[22];
    float* out = (float*)d_out;

    // Workspace layout (256B aligned offsets)
    char* ws = (char*)d_ws;
    size_t off = 0;
    auto alloc = [&](size_t bytes) -> char* {
        char* p = ws + off;
        off = (off + bytes + 255) & ~(size_t)255;
        return p;
    };
    unsigned short* act_hi = (unsigned short*)alloc((size_t)NPAD * D1 * 2); // activations hi (x or h1)
    unsigned short* act_lo = (unsigned short*)alloc((size_t)NPAD * D1 * 2); // activations lo
    unsigned short* wt_hi[4], *wt_lo[4];
    for (int i = 0; i < 4; ++i) {
        wt_hi[i] = (unsigned short*)alloc((size_t)D2 * D1 * 2);
        wt_lo[i] = (unsigned short*)alloc((size_t)D2 * D1 * 2);
    }
    float* qb  = (float*)alloc((size_t)NPAD * D2 * 4);
    float* kb  = (float*)alloc((size_t)NPAD * D2 * 4);
    float* vb  = (float*)alloc((size_t)NPAD * D2 * 4);
    float* rb  = (float*)alloc((size_t)NPAD * D2 * 4);
    float* agg = (float*)alloc((size_t)NPAD * D2 * 4);
    float* alp = (float*)alloc((size_t)EH * 4);
    float* mb  = (float*)alloc((size_t)N * 4 * 4);
    float* sb  = (float*)alloc((size_t)N * 4 * 4);
    float* gsum = (float*)alloc((size_t)G * D2 * 4);
    float* gcnt = (float*)alloc((size_t)G * 4);
    (void)ws_size; (void)n_in;

    auto blocks = [](size_t n, int b) { return (unsigned)((n + b - 1) / b); };

    // Input activations -> padded bf16 hi/lo
    {
        size_t tot = (size_t)NPAD * F0;
        cvt_pad_bf16x2<<<blocks(tot, 256), 256, 0, stream>>>(x, act_hi, act_lo, (size_t)N * F0, tot);
    }

    for (int l = 0; l < 2; ++l) {
        const int K = (l == 0) ? F0 : D1;
        const int M = (l == 0) ? D1 : D2;
        const int C = M / 4;
        const float inv_scale = 1.0f / sqrtf((float)C / 4.0f);

        // Transpose + split-convert weights: [K x M] -> bf16 hi/lo [M x K]
        const float* Ws[4] = { Wq[l], Wk[l], Wv[l], Wr[l] };
        const float* bs[4] = { bq[l], bk[l], bv[l], br[l] };
        float* Ys[4] = { qb, kb, vb, rb };
        for (int i = 0; i < 4; ++i)
            transpose_cvt_w_split<<<blocks((size_t)K * M, 256), 256, 0, stream>>>(Ws[i], wt_hi[i], wt_lo[i], K, M);

        // WMMA GEMMs (bf16x3)
        dim3 ggrid(M / 256, NPAD / 16);
        for (int i = 0; i < 4; ++i)
            gemm_bf16x3_wmma<<<ggrid, 128, 0, stream>>>(act_hi, act_lo, wt_hi[i], wt_lo[i], bs[i], Ys[i], K, M);

        // Attention scaffolding
        fill_f32<<<blocks((size_t)NPAD * M, 256), 256, 0, stream>>>(agg, 0.0f, (size_t)NPAD * M);
        fill_f32<<<blocks((size_t)N * 4, 256), 256, 0, stream>>>(mb, -INFINITY, (size_t)N * 4);
        fill_f32<<<blocks((size_t)N * 4, 256), 256, 0, stream>>>(sb, 0.0f, (size_t)N * 4);

        edge_logits<<<blocks(EH, 4), 128, 0, stream>>>(src, dst, qb, kb, alp, EH, C, inv_scale);
        seg_max<<<blocks(EH, 256), 256, 0, stream>>>(dst, alp, mb, EH);
        exp_seg_sum<<<blocks(EH, 256), 256, 0, stream>>>(dst, alp, mb, sb, EH);
        scatter_v<<<blocks(EH, 4), 128, 0, stream>>>(src, dst, vb, alp, sb, agg, EH, C);

        // h = relu(agg + root)
        add_root_relu<<<blocks((size_t)N * M, 256), 256, 0, stream>>>(agg, rb, (size_t)N * M);

        if (l == 0) {
            // h1 fp32 -> padded bf16 hi/lo for layer 2
            size_t tot = (size_t)NPAD * D1;
            cvt_pad_bf16x2<<<blocks(tot, 256), 256, 0, stream>>>(agg, act_hi, act_lo, (size_t)N * D1, tot);
        }
    }

    // Global mean pool + final linear
    fill_f32<<<blocks((size_t)G * D2, 256), 256, 0, stream>>>(gsum, 0.0f, (size_t)G * D2);
    fill_f32<<<blocks(G, 256), 256, 0, stream>>>(gcnt, 0.0f, (size_t)G);
    pool_cnt<<<blocks(N, 256), 256, 0, stream>>>(batch, gcnt, N);
    pool_sum<<<blocks((size_t)N * D2, 256), 256, 0, stream>>>(batch, agg, gsum, N, D2);
    pool_linear<<<G, 256, 0, stream>>>(gsum, gcnt, Wlin, blin, out, D2);
}
